// GaussianEncoder_39900246179896
// MI455X (gfx1250) — matrix operand order repaired and checked by reading, not compiled, not run on hardware
//
#include <hip/hip_runtime.h>
#include <hip/hip_bf16.h>
#include <math.h>

typedef __attribute__((ext_vector_type(16))) _Float16 v16h;
typedef __attribute__((ext_vector_type(8)))  float    v8f;

#define N_    512
#define F_    512
#define DK_   32
#define H_    16
#define HID1  128
#define HID2  1024

__device__ __forceinline__ float gelu_exact(float x) {
  return 0.5f * x * (1.0f + erff(x * 0.70710678118f));
}

// -------------------------------------------------------------------------
// Kernel 1: per-node projections. One wave per node.
//   gate = feat[i]·Wg + bg ; p[d] = feat[i]·Wmul[d] ; q[d] = feat[i]·Wb[d]
//   c = coords*gate ; cn = |c|^2
// -------------------------------------------------------------------------
__global__ __launch_bounds__(256) void k_node(
    const float* __restrict__ coords, const float* __restrict__ feat,
    const float* __restrict__ Wg, const float* __restrict__ bg,
    const float* __restrict__ Wmul, const float* __restrict__ Wb,
    float* __restrict__ c_ws, float* __restrict__ cn_ws,
    float* __restrict__ p_ws, float* __restrict__ q_ws)
{
  const int wid  = threadIdx.x >> 5;
  const int lane = threadIdx.x & 31;
  const int i = blockIdx.x * 8 + wid;

  float g = 0.f, p0 = 0.f, p1 = 0.f, q0 = 0.f, q1 = 0.f;
  for (int f = lane; f < F_; f += 32) {
    float v = feat[i * F_ + f];
    g  = fmaf(v, Wg[f],        g);
    p0 = fmaf(v, Wmul[f],      p0);
    p1 = fmaf(v, Wmul[F_ + f], p1);
    q0 = fmaf(v, Wb[f],        q0);
    q1 = fmaf(v, Wb[F_ + f],   q1);
  }
#pragma unroll
  for (int m = 16; m >= 1; m >>= 1) {
    g  += __shfl_xor(g,  m, 32);
    p0 += __shfl_xor(p0, m, 32);
    p1 += __shfl_xor(p1, m, 32);
    q0 += __shfl_xor(q0, m, 32);
    q1 += __shfl_xor(q1, m, 32);
  }
  if (lane == 0) {
    float gate = g + bg[0];
    float cx = coords[i*3+0] * gate;
    float cy = coords[i*3+1] * gate;
    float cz = coords[i*3+2] * gate;
    c_ws[i*3+0] = cx; c_ws[i*3+1] = cy; c_ws[i*3+2] = cz;
    cn_ws[i] = cx*cx + cy*cy + cz*cz;
    p_ws[i*2+0] = p0; p_ws[i*2+1] = p1;
    q_ws[i*2+0] = q0; q_ws[i*2+1] = q1;
  }
}

// -------------------------------------------------------------------------
// Kernel 2: per-pair gbf + attn MLP (WMMA) + centrality accumulation.
// One block per row i (4 waves); wave w covers j in [w*128, w*128+128)
// as 8 tiles of 16. Weights live in LDS (f16) shared by the block, so the
// wave stays under 256 VGPRs (no VGPR-MSB banking).
// gbf is generated straight into WMMA A-register layout:
//   lane = pair row, low lanes K{0-7,16-23}, high lanes K{8-15,24-31}.
// Layer2 computed as out^T = W2p(16x128) @ h^T(128x16) via LDS transpose.
// Centrality partials: per-lane regs -> xor-reduce -> LDS atomic -> 1 write.
// -------------------------------------------------------------------------
__global__ __launch_bounds__(128) void k_pair(
    const float* __restrict__ coords, const float* __restrict__ pair_mask,
    const float* __restrict__ bmul, const float* __restrict__ bbv,
    const float* __restrict__ means, const float* __restrict__ stds,
    const float* __restrict__ W1p, const float* __restrict__ b1p,
    const float* __restrict__ W2p, const float* __restrict__ b2p,
    const float* __restrict__ c_ws, const float* __restrict__ cn_ws,
    const float* __restrict__ p_ws, const float* __restrict__ q_ws,
    float* __restrict__ cent_ws, float* __restrict__ attn_out)
{
  __shared__ _Float16 w1sh[HID1 * DK_];     // 8 KB  W1p as f16
  __shared__ _Float16 w2sh[H_ * HID1];      // 4 KB  W2p as f16
  __shared__ _Float16 hsh[4][16 * HID1];    // 16 KB per-wave h^T staging
  __shared__ float    cent_sh[DK_];

  const int tid  = threadIdx.x;
  const int wid  = tid >> 5;
  const int lane = tid & 31;
  const int row  = lane & 15;
  const int hf   = lane >> 4;       // lane half
  const int koff = hf * 8;
  const int i    = blockIdx.x;

  // ---- stage weights into LDS once per block ----
  for (int idx = tid; idx < HID1 * DK_; idx += 128) w1sh[idx] = (_Float16)W1p[idx];
  for (int idx = tid; idx < H_ * HID1;  idx += 128) w2sh[idx] = (_Float16)W2p[idx];
  if (tid < DK_) cent_sh[tid] = 0.f;
  __syncthreads();

  // ---- per-lane gaussian constants for this lane's 16 K-slots ----
  float me[16], is[16], cf[16];
  const float inv_s2pi = 1.0f / sqrtf(2.0f * 3.14159f);
#pragma unroll
  for (int e = 0; e < 16; ++e) {
    int idx = (e < 8) ? (koff + e) : (16 + koff + (e-8));
    me[e] = means[idx];
    float s = fabsf(stds[idx]) + 1e-5f;
    is[e] = 1.0f / s;
    cf[e] = is[e] * inv_s2pi;
  }
  const float bm0 = bmul[0], bm1 = bmul[1];
  const float bb0 = bbv[0],  bb1 = bbv[1];

  // ---- i-side values (uniform across the block) ----
  const float xi = coords[i*3+0], yi = coords[i*3+1], zi = coords[i*3+2];
  const float cix = c_ws[i*3+0], ciy = c_ws[i*3+1], ciz = c_ws[i*3+2];
  const float pi0 = p_ws[i*2+0], pi1 = p_ws[i*2+1];
  const float qi0 = q_ws[i*2+0], qi1 = q_ws[i*2+1];

  float cpart[16];
#pragma unroll
  for (int e = 0; e < 16; ++e) cpart[e] = 0.f;

  for (int jt = wid * 8; jt < wid * 8 + 8; ++jt) {
    const int j = jt*16 + row;
    // geometry
    float dx = xi - coords[j*3+0], dy = yi - coords[j*3+1], dz = zi - coords[j*3+2];
    float dist = sqrtf(dx*dx + dy*dy + dz*dz + 1e-6f);
    float dotc = cix*c_ws[j*3+0] + ciy*c_ws[j*3+1] + ciz*c_ws[j*3+2];
    float product = (dotc + cn_ws[j]) * dotc;
    // x = mul*geo + bias (rank-structured edge projection)
    float x0 = (pi0 - p_ws[j*2+0] + bm0) * dist    + (qi0 - q_ws[j*2+0] + bb0);
    float x1 = (pi1 - p_ws[j*2+1] + bm1) * product + (qi1 - q_ws[j*2+1] + bb1);
    float mk = pair_mask[i*N_ + j];

    v16h ag;
#pragma unroll
    for (int e = 0; e < 16; ++e) {
      float x = (e < 8) ? x0 : x1;
      float tn = (x - me[e]) * is[e];
      float gv = __expf(-0.5f * tn * tn) * cf[e];
      ag[e] = (_Float16)gv;
      cpart[e] += gv * mk;
    }

    // layer 1: 8 WMMA tiles -> gelu -> LDS as h^T (pair-major, f16)
#pragma unroll
    for (int t = 0; t < 8; ++t) {
      v16h b1;
#pragma unroll
      for (int e = 0; e < 16; ++e)
        b1[e] = w1sh[(t*16 + row) * DK_ + ((e < 8) ? (koff + e) : (16 + koff + (e-8)))];
      v8f cc;
      float bias = b1p[t*16 + row];
#pragma unroll
      for (int v = 0; v < 8; ++v) cc[v] = bias;
      cc = __builtin_amdgcn_wmma_f32_16x16x32_f16(false, ag, false, b1,
                                                  (short)0, cc, false, false);
#pragma unroll
      for (int v = 0; v < 8; ++v) {
        float hval = gelu_exact(cc[v]);
        hsh[wid][(v + hf*8) * HID1 + t*16 + row] = (_Float16)hval;
      }
    }
    __syncthreads();

    // layer 2: out^T[16 heads x 16 pairs] = W2p @ h^T + b2p, K=128 as 4 chunks
    v8f c2;
#pragma unroll
    for (int v = 0; v < 8; ++v) c2[v] = b2p[v + hf*8];
#pragma unroll
    for (int ch = 0; ch < 4; ++ch) {
      v16h a2, bh;
#pragma unroll
      for (int e = 0; e < 16; ++e) {
        int k = (e < 8) ? (ch*32 + koff + e) : (ch*32 + 16 + koff + (e-8));
        a2[e] = w2sh[row * HID1 + k];
        bh[e] = hsh[wid][row * HID1 + k];
      }
      c2 = __builtin_amdgcn_wmma_f32_16x16x32_f16(false, a2, false, bh,
                                                  (short)0, c2, false, false);
    }
    // D: lane = pair column, VGPR v -> head v + 8*hf ; attn layout [H,N,N]
#pragma unroll
    for (int v = 0; v < 8; ++v) {
      int head = v + hf*8;
      attn_out[head * (N_*N_) + i * N_ + jt*16 + row] = c2[v];
    }
    __syncthreads();
  }

  // centrality: reduce within each 16-lane half (halves own disjoint K sets),
  // then accumulate the 4 waves' partials through LDS.
#pragma unroll
  for (int e = 0; e < 16; ++e) {
#pragma unroll
    for (int m = 1; m <= 8; m <<= 1)
      cpart[e] += __shfl_xor(cpart[e], m, 32);
  }
  if (row == 0) {
#pragma unroll
    for (int e = 0; e < 16; ++e) {
      int k = (e < 8) ? (koff + e) : (16 + koff + (e-8));
      atomicAdd(&cent_sh[k], cpart[e]);
    }
  }
  __syncthreads();
  if (tid < DK_) cent_ws[i * DK_ + tid] = cent_sh[tid];
}

// -------------------------------------------------------------------------
// Kernel 3: centrality MLP  [N,32] -> gelu(.@W1c^T+b) [N,1024] -> @W2c^T+b [N,512]
// One block per 16 nodes, 8 waves. h^T staged in 32 KB LDS.
// -------------------------------------------------------------------------
__global__ __launch_bounds__(256) void k_cent(
    const float* __restrict__ cent_ws,
    const float* __restrict__ W1c, const float* __restrict__ b1c,
    const float* __restrict__ W2c, const float* __restrict__ b2c,
    float* __restrict__ cent_out)
{
  __shared__ _Float16 hsh[16 * HID2];   // 32 KB

  const int wid  = threadIdx.x >> 5;    // 0..7
  const int lane = threadIdx.x & 31;
  const int row  = lane & 15;
  const int hf   = lane >> 4;
  const int koff = hf * 8;
  const int node0 = blockIdx.x * 16;

  // A = cent_in tile (16 nodes x 32), in WMMA A layout
  v16h ag;
#pragma unroll
  for (int e = 0; e < 16; ++e) {
    int k = (e < 8) ? (koff + e) : (16 + koff + (e-8));
    ag[e] = (_Float16)cent_ws[(node0 + row) * DK_ + k];
  }

  // layer 1: wave wid handles hidden tiles wid*8 .. wid*8+7
#pragma unroll
  for (int tt = 0; tt < 8; ++tt) {
    int t = wid * 8 + tt;
    v16h b1;
#pragma unroll
    for (int e = 0; e < 16; ++e)
      b1[e] = (_Float16)W1c[(t*16 + row) * DK_ + ((e < 8) ? (koff + e) : (16 + koff + (e-8)))];
    v8f cc;
    float bias = b1c[t*16 + row];
#pragma unroll
    for (int v = 0; v < 8; ++v) cc[v] = bias;
    cc = __builtin_amdgcn_wmma_f32_16x16x32_f16(false, ag, false, b1,
                                                (short)0, cc, false, false);
#pragma unroll
    for (int v = 0; v < 8; ++v) {
      float h = gelu_exact(cc[v]);
      hsh[(v + hf*8) * HID2 + t*16 + row] = (_Float16)h;
    }
  }
  __syncthreads();

  // layer 2: out^T = W2c(512x1024) @ h^T ; wave handles 4 out-feature tiles
#pragma unroll
  for (int tt = 0; tt < 4; ++tt) {
    int t = wid + tt * 8;
    v8f c2;
#pragma unroll
    for (int v = 0; v < 8; ++v) c2[v] = b2c[t*16 + v + hf*8];
    for (int ch = 0; ch < HID2/32; ++ch) {
      v16h a, b;
#pragma unroll
      for (int e = 0; e < 16; ++e) {
        int k = (e < 8) ? (ch*32 + koff + e) : (ch*32 + 16 + koff + (e-8));
        a[e] = (_Float16)W2c[(t*16 + row) * HID2 + k];
        b[e] = hsh[row * HID2 + k];
      }
      c2 = __builtin_amdgcn_wmma_f32_16x16x32_f16(false, a, false, b,
                                                  (short)0, c2, false, false);
    }
    // D: lane = node column, VGPR v -> out-feature t*16 + v + 8*hf
#pragma unroll
    for (int v = 0; v < 8; ++v)
      cent_out[(node0 + row) * F_ + t*16 + v + hf*8] = c2[v];
  }
}

// -------------------------------------------------------------------------
extern "C" void kernel_launch(void* const* d_in, const int* in_sizes, int n_in,
                              void* d_out, int out_size, void* d_ws, size_t ws_size,
                              hipStream_t stream) {
  const float* coords = (const float*)d_in[0];
  const float* feat   = (const float*)d_in[1];
  const float* mask   = (const float*)d_in[2];
  const float* Wg     = (const float*)d_in[3];
  const float* bg     = (const float*)d_in[4];
  const float* Wmul   = (const float*)d_in[5];
  const float* bmul   = (const float*)d_in[6];
  const float* Wb     = (const float*)d_in[7];
  const float* bb     = (const float*)d_in[8];
  const float* means  = (const float*)d_in[9];
  const float* stds   = (const float*)d_in[10];
  const float* W1p    = (const float*)d_in[11];
  const float* b1p    = (const float*)d_in[12];
  const float* W2p    = (const float*)d_in[13];
  const float* b2p    = (const float*)d_in[14];
  const float* W1c    = (const float*)d_in[15];
  const float* b1c    = (const float*)d_in[16];
  const float* W2c    = (const float*)d_in[17];
  const float* b2c    = (const float*)d_in[18];

  float* ws      = (float*)d_ws;
  float* c_ws    = ws;             // 512*3
  float* cn_ws   = ws + 1536;      // 512
  float* p_ws    = ws + 2048;      // 512*2
  float* q_ws    = ws + 3072;      // 512*2
  float* cent_ws = ws + 4096;      // 512*32

  float* cent_out = (float*)d_out;              // [N, F] = 262144 floats
  float* attn_out = cent_out + N_ * F_;         // [H, N, N] = 4194304 floats

  k_node<<<64, 256, 0, stream>>>(coords, feat, Wg, bg, Wmul, Wb,
                                 c_ws, cn_ws, p_ws, q_ws);
  k_pair<<<512, 128, 0, stream>>>(coords, mask, bmul, bb, means, stds,
                                  W1p, b1p, W2p, b2p,
                                  c_ws, cn_ws, p_ws, q_ws,
                                  cent_ws, attn_out);
  k_cent<<<32, 256, 0, stream>>>(cent_ws, W1c, b1c, W2c, b2c, cent_out);
}
